// Network_90606630076529
// MI455X (gfx1250) — matrix operand order repaired and checked by reading, not compiled, not run on hardware
//
#include <hip/hip_runtime.h>

typedef __attribute__((ext_vector_type(16))) _Float16 v16h;
typedef __attribute__((ext_vector_type(8)))  _Float16 v8h;
typedef __attribute__((ext_vector_type(8)))  float    v8f;

#define B_ 4

// ---------------------------------------------------------------------------
// Furthest point sampling: one block per batch, sequential npoint-1 scan.
// Running min-distances live in registers (<=8 per thread), argmax via LDS tree.
// ---------------------------------------------------------------------------
__global__ void fps_kernel(const float* __restrict__ xyz, int N, int M,
                           int* __restrict__ outIdx)
{
    __shared__ float sval[1024];
    __shared__ int   sidx[1024];
    __shared__ float lastP[3];
    int b = blockIdx.x;
    int t = threadIdx.x;
    const float* pts = xyz + (size_t)b * N * 3;
    int* idx = outIdx + (size_t)b * M;
    float dmin[8];
#pragma unroll
    for (int i = 0; i < 8; ++i) dmin[i] = 1e10f;
    if (t == 0) { idx[0] = 0; lastP[0] = pts[0]; lastP[1] = pts[1]; lastP[2] = pts[2]; }
    __syncthreads();
    for (int s = 1; s < M; ++s) {
        float lx = lastP[0], ly = lastP[1], lz = lastP[2];
        float best = -1.0f; int bidx = 0x7fffffff;
#pragma unroll
        for (int i = 0; i < 8; ++i) {
            int p = t + i * 1024;
            if (p < N) {
                float dx = pts[p*3+0] - lx;
                float dy = pts[p*3+1] - ly;
                float dz = pts[p*3+2] - lz;
                float d = dx*dx + dy*dy + dz*dz;
                if (d < dmin[i]) dmin[i] = d;
                if (dmin[i] > best || (dmin[i] == best && p < bidx)) { best = dmin[i]; bidx = p; }
            }
        }
        sval[t] = best; sidx[t] = bidx;
        __syncthreads();
        for (int off = 512; off > 0; off >>= 1) {
            if (t < off) {
                float v2 = sval[t+off]; int i2 = sidx[t+off];
                if (v2 > sval[t] || (v2 == sval[t] && i2 < sidx[t])) { sval[t] = v2; sidx[t] = i2; }
            }
            __syncthreads();
        }
        if (t == 0) {
            int nx = sidx[0];
            idx[s] = nx;
            lastP[0] = pts[nx*3+0]; lastP[1] = pts[nx*3+1]; lastP[2] = pts[nx*3+2];
        }
        __syncthreads();
    }
}

// ---------------------------------------------------------------------------
__global__ void gather_xyz_kernel(const float* __restrict__ xyz, const int* __restrict__ idx,
                                  int N, int M, float* __restrict__ out)
{
    int gid = blockIdx.x * blockDim.x + threadIdx.x;
    if (gid >= B_ * M) return;
    int b = gid / M;
    int src = idx[gid];
    const float* p = xyz + ((size_t)b * N + src) * 3;
    float* o = out + (size_t)gid * 3;
    o[0] = p[0]; o[1] = p[1]; o[2] = p[2];
}

// ---------------------------------------------------------------------------
// 32-NN: one thread per query, insertion into sorted list; reference points
// are L2-resident (<=96KB/batch) so repeated streaming is cheap.
// ---------------------------------------------------------------------------
__global__ void knn32_kernel(const float* __restrict__ q, const float* __restrict__ r,
                             int M, int N, int* __restrict__ outIdx)
{
    const int K = 32;
    int gid = blockIdx.x * blockDim.x + threadIdx.x;
    if (gid >= B_ * M) return;
    int b = gid / M;
    const float* qp = q + (size_t)gid * 3;
    const float* rp = r + (size_t)b * N * 3;
    float qx = qp[0], qy = qp[1], qz = qp[2];
    float dk[32]; int ik[32];
    for (int i = 0; i < K; ++i) { dk[i] = 3.0e38f; ik[i] = 0; }
    for (int n = 0; n < N; ++n) {
        float dx = rp[n*3+0]-qx, dy = rp[n*3+1]-qy, dz = rp[n*3+2]-qz;
        float d = dx*dx + dy*dy + dz*dz;
        if (d < dk[K-1]) {
            int j = K-1;
            while (j > 0 && dk[j-1] > d) { dk[j] = dk[j-1]; ik[j] = ik[j-1]; --j; }
            dk[j] = d; ik[j] = n;
        }
    }
    int* o = outIdx + (size_t)gid * K;
    for (int i = 0; i < K; ++i) o[i] = ik[i];
}

// ---------------------------------------------------------------------------
// Build grouped SA input: X[(b*M+m)*K+k][CinPad] = [rel_xyz(3) | edge(Cf) | 0-pad]
// feats==nullptr means level-0 (features are the coords -> edge == rel).
// ---------------------------------------------------------------------------
__global__ void sa_group_kernel(const float* __restrict__ xyz, const float* __restrict__ nxyz,
                                const int* __restrict__ cidx, const int* __restrict__ nidx,
                                const _Float16* __restrict__ feats,
                                int N, int M, int Cf, int CinPad,
                                _Float16* __restrict__ X)
{
    const int K = 32;
    int gid = blockIdx.x * blockDim.x + threadIdx.x;
    if (gid >= B_ * M * K) return;
    int bm = gid / K;
    int b = bm / M;
    int n = nidx[gid];
    const float* gp = xyz + ((size_t)b * N + n) * 3;
    const float* cp = nxyz + (size_t)bm * 3;
    _Float16* row = X + (size_t)gid * CinPad;
    float rx = gp[0]-cp[0], ry = gp[1]-cp[1], rz = gp[2]-cp[2];
    row[0] = (_Float16)rx; row[1] = (_Float16)ry; row[2] = (_Float16)rz;
    int c;
    if (feats) {
        int ci = cidx[bm];
        const _Float16* gf = feats + ((size_t)b * N + n)  * Cf;
        const _Float16* cf = feats + ((size_t)b * N + ci) * Cf;
        for (int j = 0; j < Cf; ++j)
            row[3 + j] = (_Float16)((float)gf[j] - (float)cf[j]);
        c = 3 + Cf;
    } else {
        row[3] = (_Float16)rx; row[4] = (_Float16)ry; row[5] = (_Float16)rz;
        c = 6;
    }
    for (; c < CinPad; ++c) row[c] = (_Float16)0.f;
}

// ---------------------------------------------------------------------------
// FP module front-end: 3-NN inverse-distance interpolation + concat of skip
// features, written as the f16 pixel-major GEMM operand.
// ---------------------------------------------------------------------------
__global__ void fp_build_kernel(const float* __restrict__ uxyz, const float* __restrict__ kxyz,
                                const _Float16* __restrict__ kfeats, int Ck,
                                const _Float16* __restrict__ ufeats, int Cu,
                                int Nu, int Nk, int CinPad,
                                _Float16* __restrict__ X)
{
    int gid = blockIdx.x * blockDim.x + threadIdx.x;
    if (gid >= B_ * Nu) return;
    int b = gid / Nu;
    const float* up = uxyz + (size_t)gid * 3;
    const float* rp = kxyz + (size_t)b * Nk * 3;
    float qx = up[0], qy = up[1], qz = up[2];
    float d0 = 3e38f, d1 = 3e38f, d2 = 3e38f;
    int j0 = 0, j1 = 0, j2 = 0;
    for (int n = 0; n < Nk; ++n) {
        float dx = rp[n*3+0]-qx, dy = rp[n*3+1]-qy, dz = rp[n*3+2]-qz;
        float d = dx*dx + dy*dy + dz*dz;
        if (d < d0)      { d2 = d1; j2 = j1; d1 = d0; j1 = j0; d0 = d; j0 = n; }
        else if (d < d1) { d2 = d1; j2 = j1; d1 = d; j1 = n; }
        else if (d < d2) { d2 = d; j2 = n; }
    }
    float w0 = 1.f/(d0+1e-8f), w1 = 1.f/(d1+1e-8f), w2 = 1.f/(d2+1e-8f);
    float wsum = w0 + w1 + w2;
    w0 /= wsum; w1 /= wsum; w2 /= wsum;
    const _Float16* f0 = kfeats + ((size_t)b*Nk + j0) * Ck;
    const _Float16* f1 = kfeats + ((size_t)b*Nk + j1) * Ck;
    const _Float16* f2 = kfeats + ((size_t)b*Nk + j2) * Ck;
    _Float16* row = X + (size_t)gid * CinPad;
    for (int c = 0; c < Ck; ++c)
        row[c] = (_Float16)(w0*(float)f0[c] + w1*(float)f1[c] + w2*(float)f2[c]);
    int c = Ck;
    if (ufeats) {
        const _Float16* uf = ufeats + (size_t)gid * Cu;
        for (int j = 0; j < Cu; ++j) row[c + j] = uf[j];
        c += Cu;
    } else {
        row[c+0] = (_Float16)qx; row[c+1] = (_Float16)qy; row[c+2] = (_Float16)qz;
        c += 3;
    }
    for (; c < CinPad; ++c) row[c] = (_Float16)0.f;
}

// ---------------------------------------------------------------------------
// WMMA GEMM: Y[p][co] = relu(sum_ci W[co][ci]*X[p][ci] + b[co])
// One wave -> 16 Cout x 64 pixels (4 accumulators reusing the A tile).
// A k order used here (contiguous, not AMD's table order): lane<16 -> K 0..15,
// lane>=16 -> K 16..31.  B uses the same order: lane<16 col=lane K 0..15,
// lane>=16 col=lane-16 K 16..31 -> one contiguous v16h per tile from
// pixel-major X.  C/D: vgpr g -> row g+(lane>=16?8:0), col lane&15.
// mode 0: f16 pixel-major out (feeds next layer). mode 1: f32 [B][Cout][Npts].
// ---------------------------------------------------------------------------
__global__ void mlp_gemm_kernel(const _Float16* __restrict__ W,
                                const float* __restrict__ bias,
                                const _Float16* __restrict__ X,
                                _Float16* __restrict__ Yh,
                                float* __restrict__ Yf,
                                int CinPad, int Cout, int P, int Npts, int mode)
{
    int wave = blockIdx.x * (blockDim.x >> 5) + (threadIdx.x >> 5);
    int lane = threadIdx.x & 31;
    int tilesP = P >> 6;
    int tilesC = Cout >> 4;
    if (wave >= tilesP * tilesC) return;
    int ct = wave / tilesP;
    int pt = wave % tilesP;

    int hi = (lane & 16) ? 1 : 0;
    int rowA = ct * 16 + (lane & 15);
    const _Float16* wrow = W + (size_t)rowA * CinPad + (hi ? 16 : 0);
    int colBase = pt * 64 + (lane & 15);
    int hioff = hi ? 16 : 0;

    v8f zero = {0.f,0.f,0.f,0.f,0.f,0.f,0.f,0.f};
    v8f acc[4];
#pragma unroll
    for (int s = 0; s < 4; ++s) acc[s] = zero;

    for (int kk = 0; kk < CinPad; kk += 32) {
        v8h a0 = *(const v8h*)(wrow + kk);
        v8h a1 = *(const v8h*)(wrow + kk + 8);
        v16h A;
#pragma unroll
        for (int i = 0; i < 8; ++i) { A[i] = a0[i]; A[i+8] = a1[i]; }
#pragma unroll
        for (int s = 0; s < 4; ++s) {
            const _Float16* xp = X + (size_t)(colBase + s*16) * CinPad + kk + hioff;
            v16h Bm = *(const v16h*)xp;
            acc[s] = __builtin_amdgcn_wmma_f32_16x16x32_f16(
                         false, A, false, Bm, (short)0, acc[s], false, false);
        }
    }

    int coBase = ct * 16 + (hi ? 8 : 0);
    float bv[8];
#pragma unroll
    for (int g = 0; g < 8; ++g) bv[g] = bias[coBase + g];

    if (mode == 0) {
#pragma unroll
        for (int s = 0; s < 4; ++s) {
            int p = colBase + s * 16;
            v8h o;
#pragma unroll
            for (int g = 0; g < 8; ++g) {
                float v = acc[s][g] + bv[g];
                o[g] = (_Float16)(v > 0.f ? v : 0.f);
            }
            *(v8h*)(Yh + (size_t)p * Cout + coBase) = o;
        }
    } else {
#pragma unroll
        for (int s = 0; s < 4; ++s) {
            int p = colBase + s * 16;
            int bb = p / Npts;
            int n = p - bb * Npts;
            float* yr = Yf + ((size_t)bb * Cout + coBase) * Npts + n;
#pragma unroll
            for (int g = 0; g < 8; ++g) {
                float v = acc[s][g] + bv[g];
                yr[(size_t)g * Npts] = (v > 0.f ? v : 0.f);
            }
        }
    }
}

// ---------------------------------------------------------------------------
__global__ void maxpool_kernel(const _Float16* __restrict__ Y, int C, int nPts,
                               _Float16* __restrict__ F)
{
    const int K = 32;
    int gid = blockIdx.x * blockDim.x + threadIdx.x;
    if (gid >= nPts * C) return;
    int pt = gid / C, c = gid - pt * C;
    const _Float16* yp = Y + ((size_t)pt * K) * C + c;
    float m = -3.0e38f;
    for (int k = 0; k < K; ++k) {
        float v = (float)yp[(size_t)k * C];
        if (v > m) m = v;
    }
    F[gid] = (_Float16)m;
}

// ---------------------------------------------------------------------------
__global__ void wconv_kernel(const float* __restrict__ W, int Cin, int CinPad, int total,
                             _Float16* __restrict__ W16)
{
    int gid = blockIdx.x * blockDim.x + threadIdx.x;
    if (gid >= total) return;
    int co = gid / CinPad, ci = gid - co * CinPad;
    W16[gid] = (ci < Cin) ? (_Float16)W[(size_t)co * Cin + ci] : (_Float16)0.f;
}

// ---------------------------------------------------------------------------
extern "C" void kernel_launch(void* const* d_in, const int* in_sizes, int n_in,
                              void* d_out, int out_size, void* d_ws, size_t ws_size,
                              hipStream_t stream)
{
    (void)in_sizes; (void)n_in; (void)out_size; (void)ws_size;
    const float* pc = (const float*)d_in[0];

    struct LayerDesc { const float* W; const float* b; int Cin, CinPad, Cout; _Float16* W16; };
    LayerDesc L[16] = {
        { (const float*)d_in[1],  (const float*)d_in[2],  6,   32,  32,  nullptr }, // sa0
        { (const float*)d_in[3],  (const float*)d_in[4],  32,  32,  32,  nullptr },
        { (const float*)d_in[5],  (const float*)d_in[6],  32,  32,  64,  nullptr },
        { (const float*)d_in[7],  (const float*)d_in[8],  67,  96,  64,  nullptr }, // sa1
        { (const float*)d_in[9],  (const float*)d_in[10], 64,  64,  64,  nullptr },
        { (const float*)d_in[11], (const float*)d_in[12], 64,  64,  256, nullptr },
        { (const float*)d_in[13], (const float*)d_in[14], 259, 288, 256, nullptr }, // sa2
        { (const float*)d_in[15], (const float*)d_in[16], 256, 256, 256, nullptr },
        { (const float*)d_in[17], (const float*)d_in[18], 256, 256, 512, nullptr },
        { (const float*)d_in[19], (const float*)d_in[20], 259, 288, 256, nullptr }, // fp0
        { (const float*)d_in[21], (const float*)d_in[22], 256, 256, 256, nullptr },
        { (const float*)d_in[23], (const float*)d_in[24], 256, 256, 128, nullptr },
        { (const float*)d_in[25], (const float*)d_in[26], 576, 576, 256, nullptr }, // fp1
        { (const float*)d_in[27], (const float*)d_in[28], 256, 256, 256, nullptr },
        { (const float*)d_in[29], (const float*)d_in[30], 768, 768, 256, nullptr }, // fp2
        { (const float*)d_in[31], (const float*)d_in[32], 256, 256, 512, nullptr },
    };

    char* ws = (char*)d_ws;
    size_t off = 0;
    auto alloc = [&](size_t bytes) -> void* {
        size_t o = (off + 255) & ~(size_t)255;
        off = o + bytes;
        return (void*)(ws + o);
    };

    for (int i = 0; i < 16; ++i)
        L[i].W16 = (_Float16*)alloc((size_t)L[i].Cout * L[i].CinPad * sizeof(_Float16));

    float* nx1 = (float*)alloc((size_t)B_*2048*3*sizeof(float));
    float* nx2 = (float*)alloc((size_t)B_*512*3*sizeof(float));
    float* nx3 = (float*)alloc((size_t)B_*128*3*sizeof(float));
    int* cidx0 = (int*)alloc((size_t)B_*2048*sizeof(int));
    int* cidx1 = (int*)alloc((size_t)B_*512*sizeof(int));
    int* cidx2 = (int*)alloc((size_t)B_*128*sizeof(int));
    int* nidx0 = (int*)alloc((size_t)B_*2048*32*sizeof(int));
    int* nidx1 = (int*)alloc((size_t)B_*512*32*sizeof(int));
    int* nidx2 = (int*)alloc((size_t)B_*128*32*sizeof(int));
    _Float16* feats1 = (_Float16*)alloc((size_t)B_*2048*64*sizeof(_Float16));
    _Float16* feats2 = (_Float16*)alloc((size_t)B_*512*256*sizeof(_Float16));
    _Float16* feats3 = (_Float16*)alloc((size_t)B_*128*512*sizeof(_Float16));
    _Float16* f2buf  = (_Float16*)alloc((size_t)B_*512*512*sizeof(_Float16));
    _Float16* f1buf  = (_Float16*)alloc((size_t)B_*2048*256*sizeof(_Float16));
    const size_t BIG = (size_t)262144 * 64; // halves; largest activation tensor
    _Float16* bufA = (_Float16*)alloc(BIG * sizeof(_Float16));
    _Float16* bufB = (_Float16*)alloc(BIG * sizeof(_Float16));

    auto grid1 = [](int total, int bs){ return (total + bs - 1) / bs; };

    for (int i = 0; i < 16; ++i) {
        int total = L[i].Cout * L[i].CinPad;
        wconv_kernel<<<grid1(total,256),256,0,stream>>>(L[i].W, L[i].Cin, L[i].CinPad, total, L[i].W16);
    }

    auto gemm = [&](int li, const _Float16* Xin, int P, _Float16* Yh, float* Yf, int Npts){
        int tiles = (L[li].Cout/16) * (P/64);
        mlp_gemm_kernel<<<grid1(tiles,8),256,0,stream>>>(L[li].W16, L[li].b, Xin, Yh, Yf,
            L[li].CinPad, L[li].Cout, P, Npts, Yf ? 1 : 0);
    };

    // ---- SA stage 0: 8192 -> 2048 pts, [6,32,32,64] ----
    fps_kernel<<<B_,1024,0,stream>>>(pc, 8192, 2048, cidx0);
    gather_xyz_kernel<<<grid1(B_*2048,256),256,0,stream>>>(pc, cidx0, 8192, 2048, nx1);
    knn32_kernel<<<grid1(B_*2048,256),256,0,stream>>>(nx1, pc, 2048, 8192, nidx0);
    sa_group_kernel<<<grid1(B_*2048*32,256),256,0,stream>>>(pc, nx1, cidx0, nidx0, nullptr,
                                                            8192, 2048, 3, 32, bufA);
    gemm(0, bufA, B_*2048*32, bufB, nullptr, 0);
    gemm(1, bufB, B_*2048*32, bufA, nullptr, 0);
    gemm(2, bufA, B_*2048*32, bufB, nullptr, 0);
    maxpool_kernel<<<grid1(B_*2048*64,256),256,0,stream>>>(bufB, 64, B_*2048, feats1);

    // ---- SA stage 1: 2048 -> 512 pts, [67,64,64,256] ----
    fps_kernel<<<B_,1024,0,stream>>>(nx1, 2048, 512, cidx1);
    gather_xyz_kernel<<<grid1(B_*512,256),256,0,stream>>>(nx1, cidx1, 2048, 512, nx2);
    knn32_kernel<<<grid1(B_*512,256),256,0,stream>>>(nx2, nx1, 512, 2048, nidx1);
    sa_group_kernel<<<grid1(B_*512*32,256),256,0,stream>>>(nx1, nx2, cidx1, nidx1, feats1,
                                                           2048, 512, 64, 96, bufA);
    gemm(3, bufA, B_*512*32, bufB, nullptr, 0);
    gemm(4, bufB, B_*512*32, bufA, nullptr, 0);
    gemm(5, bufA, B_*512*32, bufB, nullptr, 0);
    maxpool_kernel<<<grid1(B_*512*256,256),256,0,stream>>>(bufB, 256, B_*512, feats2);

    // ---- SA stage 2: 512 -> 128 pts, [259,256,256,512] ----
    fps_kernel<<<B_,1024,0,stream>>>(nx2, 512, 128, cidx2);
    gather_xyz_kernel<<<grid1(B_*128,256),256,0,stream>>>(nx2, cidx2, 512, 128, nx3);
    knn32_kernel<<<grid1(B_*128,256),256,0,stream>>>(nx3, nx2, 128, 512, nidx2);
    sa_group_kernel<<<grid1(B_*128*32,256),256,0,stream>>>(nx2, nx3, cidx2, nidx2, feats2,
                                                           512, 128, 256, 288, bufA);
    gemm(6, bufA, B_*128*32, bufB, nullptr, 0);
    gemm(7, bufB, B_*128*32, bufA, nullptr, 0);
    gemm(8, bufA, B_*128*32, bufB, nullptr, 0);
    maxpool_kernel<<<grid1(B_*128*512,256),256,0,stream>>>(bufB, 512, B_*128, feats3);

    // ---- FP2: interp 128->512 pts, Cin = 512+256 = 768, [768,256,512] ----
    fp_build_kernel<<<grid1(B_*512,256),256,0,stream>>>(nx2, nx3, feats3, 512, feats2, 256,
                                                        512, 128, 768, bufA);
    gemm(14, bufA, B_*512, bufB, nullptr, 0);
    gemm(15, bufB, B_*512, f2buf, nullptr, 0);

    // ---- FP1: interp 512->2048 pts, Cin = 512+64 = 576, [576,256,256] ----
    fp_build_kernel<<<grid1(B_*2048,256),256,0,stream>>>(nx1, nx2, f2buf, 512, feats1, 64,
                                                         2048, 512, 576, bufA);
    gemm(12, bufA, B_*2048, bufB, nullptr, 0);
    gemm(13, bufB, B_*2048, f1buf, nullptr, 0);

    // ---- FP0: interp 2048->8192 pts, Cin = 256+3 = 259, [259,256,256,128] ----
    fp_build_kernel<<<grid1(B_*8192,256),256,0,stream>>>(pc, nx1, f1buf, 256, nullptr, 3,
                                                         8192, 2048, 288, bufA);
    gemm(9,  bufA, B_*8192, bufB, nullptr, 0);
    gemm(10, bufB, B_*8192, bufA, nullptr, 0);
    gemm(11, bufA, B_*8192, nullptr, (float*)d_out, 8192); // f32 (B,128,8192) out
}
